// xgnn_poly_vectorial_23313082483598
// MI455X (gfx1250) — compile-verified, matrix operand order checked by reading, not run on hardware
//
#include <hip/hip_runtime.h>
#include <hip/hip_bf16.h>
#include <math.h>

// ---------------- types ----------------
typedef __attribute__((ext_vector_type(16))) _Float16 v16h;
typedef __attribute__((ext_vector_type(8)))  _Float16 v8h;
typedef __attribute__((ext_vector_type(8)))  float    v8f;
typedef __attribute__((ext_vector_type(4)))  int      v4i;

#define NATOMS 10000
#define DEG    6
#define NEDGE  (NATOMS*DEG)      // 60000
#define NTRIP  (NEDGE*DEG)       // 360000
#define CDIM   256
#define EMBD   128
#define NRBF   16
#define NSBF   7

__device__ __forceinline__ float silu_f(float x) {
    return x * (1.0f / (1.0f + __expf(-x)));
}

// ---- optional gfx1250 async global->LDS copy path (guarded; falls back to VALU staging) ----
// Builtin signature (from probe): void(v4i32 AS1*, v4i32 AS3*, Imm int offset, Imm int cpol)
#if __has_builtin(__builtin_amdgcn_global_load_async_to_lds_b128) && \
    __has_builtin(__builtin_amdgcn_s_wait_asynccnt)
#define HAVE_ASYNC_LDS 1
typedef __attribute__((address_space(1))) v4i gv4i;
typedef __attribute__((address_space(3))) v4i lv4i;
__device__ __forceinline__ void async_ld_b128(const void* gsrc, void* ldst) {
    __builtin_amdgcn_global_load_async_to_lds_b128((gv4i*)gsrc, (lv4i*)ldst, 0, 0);
}
__device__ __forceinline__ void async_wait0() {
    __builtin_amdgcn_s_wait_asynccnt(0);
}
#else
#define HAVE_ASYNC_LDS 0
#endif

// ---------------- weight packing into WMMA B-fragment layout ----------------
// lanes 0-15 hold kin {0..7,16..23}, lanes 16-31 hold kin {8..15,24..31}, 2 halfs/VGPR.
// Packed so each lane loads 16 contiguous f16 (32B) per 16-col tile per 32-k step.
__global__ __launch_bounds__(256) void pack_weights_kernel(const float* __restrict__ W,
                                                           _Float16* __restrict__ dst,
                                                           int K, int KPAD)
{
    int idx = blockIdx.x * 256 + threadIdx.x;
    if (idx >= KPAD * CDIM) return;
    int k   = idx >> 8;
    int col = idx & 255;
    float v = (k < K) ? W[(size_t)k * CDIM + col] : 0.0f;
    int kblk = k >> 5, kin = k & 31;
    int lane = (col & 15) | (((kin >> 3) & 1) << 4);
    int vreg = ((kin & 16) >> 2) + ((kin & 7) >> 1);
    int elem = 2 * vreg + (kin & 1);
    int nt   = col >> 4;
    dst[(size_t)(((kblk * 16 + nt) * 32 + lane) * 16 + elem)] = (_Float16)v;
}

// ---------------- bond geometry: env, bond_vec, node_rbf ----------------
__global__ __launch_bounds__(256) void geom_kernel(const float* __restrict__ pos,
                                                   const int* __restrict__ ei,
                                                   float* __restrict__ bvec,
                                                   float* __restrict__ env,
                                                   float* __restrict__ nrbf)
{
    int e = blockIdx.x * 256 + threadIdx.x;
    if (e >= NEDGE) return;
    int s = ei[e], d = ei[NEDGE + e];
    float bx = pos[s*3+0] - pos[d*3+0];
    float by = pos[s*3+1] - pos[d*3+1];
    float bz = pos[s*3+2] - pos[d*3+2];
    float dist = sqrtf(bx*bx + by*by + bz*bz);
    float u  = dist * 0.2f;
    float u5 = u*u*u*u*u;
    float ev = (dist < 5.0f) ? (1.0f - 21.0f*u5 + 35.0f*u5*u - 15.0f*u5*u*u) : 0.0f;
    bvec[e*3+0] = bx; bvec[e*3+1] = by; bvec[e*3+2] = bz;
    env[e] = ev;
    float inv = ev * 0.63245553203f / fmaxf(dist, 1e-9f);   // sqrt(2/5)/d * env
    float w = dist * 0.62831853072f;                         // pi*d/5
    #pragma unroll
    for (int n = 1; n <= NRBF; ++n)
        nrbf[(size_t)e * NRBF + (n - 1)] = inv * __sinf((float)n * w);
}

// ---------------- rbf gate: grbf = silu(node_rbf @ rbf_W + rbf_b), K=16 ----------------
__global__ __launch_bounds__(256) void grbf_kernel(const float* __restrict__ nrbf,
                                                   const float* __restrict__ rbfW,
                                                   const float* __restrict__ rbfb,
                                                   _Float16* __restrict__ grbf)
{
    int idx = blockIdx.x * 256 + threadIdx.x;
    if (idx >= NEDGE * CDIM) return;
    int e = idx >> 8, c = idx & 255;
    float acc = rbfb[c];
    #pragma unroll
    for (int r = 0; r < NRBF; ++r)
        acc += nrbf[(size_t)e * NRBF + r] * rbfW[r * CDIM + c];
    grbf[(size_t)e * CDIM + c] = (_Float16)silu_f(acc);
}

// ---------------- triplet angles ----------------
__global__ __launch_bounds__(256) void theta_kernel(const float* __restrict__ pos,
                                                    const int* __restrict__ ai,
                                                    const int* __restrict__ aj,
                                                    const int* __restrict__ ak,
                                                    float* __restrict__ theta)
{
    int t = blockIdx.x * 256 + threadIdx.x;
    if (t >= NTRIP) return;
    int i = ai[t], j = aj[t], k = ak[t];
    float jx = pos[i*3+0]-pos[j*3+0], jy = pos[i*3+1]-pos[j*3+1], jz = pos[i*3+2]-pos[j*3+2];
    float kx = pos[k*3+0]-pos[j*3+0], ky = pos[k*3+1]-pos[j*3+1], kz = pos[k*3+2]-pos[j*3+2];
    float cx = jy*kz - jz*ky, cy = jz*kx - jx*kz, cz = jx*ky - jy*kx;
    float sn = sqrtf(cx*cx + cy*cy + cz*cz);
    float cs = jx*kx + jy*ky + jz*kz;
    theta[t] = atan2f(sn, cs);
}

// ---------------- per-edge scalar s = h . out_W + out_b (wave-per-edge) ----------------
__global__ __launch_bounds__(256) void s_kernel(const float* __restrict__ h,
                                                const float* __restrict__ outW,
                                                const float* __restrict__ outb,
                                                float* __restrict__ sbuf)
{
    int wave = threadIdx.x >> 5;
    int lane = threadIdx.x & 31;
    int e = blockIdx.x * 8 + wave;
    if (e >= NEDGE) return;
    const float* hr = h + (size_t)e * CDIM;
    float acc = 0.0f;
    #pragma unroll
    for (int i = 0; i < 8; ++i)
        acc += hr[lane + 32 * i] * outW[lane + 32 * i];
    #pragma unroll
    for (int off = 16; off > 0; off >>= 1)
        acc += __shfl_down(acc, off, 32);
    if (lane == 0) sbuf[e] = acc + outb[0];
}

// ---------------- final displacement (edges for atom n are 6n..6n+5) ----------------
__global__ __launch_bounds__(256) void out_kernel(const float* __restrict__ pos,
                                                  const float* __restrict__ sbuf,
                                                  const float* __restrict__ bvec,
                                                  float* __restrict__ out)
{
    int n = blockIdx.x * 256 + threadIdx.x;
    if (n >= NATOMS) return;
    float ax = 0.f, ay = 0.f, az = 0.f;
    #pragma unroll
    for (int d = 0; d < DEG; ++d) {
        int e = n * DEG + d;
        float sv = sbuf[e];
        ax += sv * bvec[e*3+0];
        ay += sv * bvec[e*3+1];
        az += sv * bvec[e*3+2];
    }
    out[n*3+0] = pos[n*3+0] + ax;
    out[n*3+1] = pos[n*3+1] + ay;
    out[n*3+2] = pos[n*3+2] + az;
}

// ---------------- shared GEMM argument block ----------------
enum { MODE_NEO1 = 0, MODE_NEO2, MODE_SBF, MODE_UPD, MODE_ATTR };

struct GemmArgs {
    const _Float16* Wp;        // packed f16 weights [KPAD/32][16][32][16]
    const float*    bias;      // [256]
    int             nrows;
    const float*    edge_attr; // NEO1 [E,338]
    const float*    env;       // NEO1 [E]
    const _Float16* neo1;      // NEO2 input [E,256]
    const _Float16* grbf;      // NEO2 gate  [E,256]
    float*          h;         // NEO2 out / MSG gather src / UPD in-out [E,256] f32
    const int*      src_e;     // MSG,SBF [T]
    const int*      atom_j;    // MSG [T]
    const _Float16* sbf_g;     // MSG [T,256]
    const _Float16* attr_atom; // MSG [N,256]
    const float*    theta;     // SBF [T]
    const float*    node_rbf;  // SBF [E,16]
    const _Float16* aggin;     // UPD [E,256]
    const float*    emb;       // ATTR [95,128]
    const int*      xz;        // ATTR [N]
    _Float16*       outh;      // f16 output
};

// ---------------- generic fused WMMA GEMM (128 rows/block, 8 waves) ----------------
template <int MODE, int KPAD>
__global__ __launch_bounds__(256) void gemm_wmma_kernel(GemmArgs g)
{
    __shared__ _Float16 As[128 * 32] __attribute__((aligned(32)));   // 8 KB

    const int tid  = threadIdx.x;
    const int lane = tid & 31;
    const int wave = tid >> 5;
    const int r0   = blockIdx.x * 128;

    v8f acc[16];
    {
        v8f z = {};
        #pragma unroll
        for (int n = 0; n < 16; ++n) acc[n] = z;
    }

    // staging assignment: each thread fills 16 halfs of one half-row
    const int  srow   = tid >> 1;
    const int  seg    = (tid & 1) << 4;
    const int  rowi   = r0 + srow;
    const bool rvalid = rowi < g.nrows;
    const int  trc    = rvalid ? rowi : (g.nrows - 1);

    // hoist per-row scalars out of the K loop
    float rEnv = 0.0f, rTheta = 0.0f;
    int   rSrc = 0;
    if (MODE == MODE_NEO1) rEnv = g.env[trc];
    if (MODE == MODE_SBF)  { rTheta = g.theta[trc]; rSrc = g.src_e[trc]; }
    int   rAtom = (MODE == MODE_ATTR) ? g.xz[trc] : 0;

    for (int ks = 0; ks < KPAD / 32; ++ks) {
        __syncthreads();
        const int kbase = ks * 32 + seg;

        if ((MODE == MODE_NEO2 || MODE == MODE_UPD) && HAVE_ASYNC_LDS) {
#if HAVE_ASYNC_LDS
            const _Float16* src =
                (MODE == MODE_NEO2 ? g.neo1 : g.aggin) + (size_t)trc * CDIM + kbase;
            async_ld_b128(src,     &As[srow * 32 + seg]);
            async_ld_b128(src + 8, &As[srow * 32 + seg + 8]);
            async_wait0();
#endif
        } else {
            float vals[16];
            if (MODE == MODE_NEO1) {
                #pragma unroll
                for (int j = 0; j < 16; ++j) {
                    int k = kbase + j;
                    vals[j] = (k < 338) ? g.edge_attr[(size_t)trc * 338 + k] * rEnv : 0.0f;
                }
            } else if (MODE == MODE_NEO2) {
                #pragma unroll
                for (int j = 0; j < 16; ++j)
                    vals[j] = (float)g.neo1[(size_t)trc * CDIM + kbase + j];
            } else if (MODE == MODE_SBF) {
                int s = kbase >> 4;              // 16-aligned chunk => one angular order
                if (s < NSBF) {
                    float c = __cosf((float)s * rTheta);
                    #pragma unroll
                    for (int j = 0; j < 16; ++j)
                        vals[j] = c * g.node_rbf[(size_t)rSrc * NRBF + j];
                } else {
                    #pragma unroll
                    for (int j = 0; j < 16; ++j) vals[j] = 0.0f;
                }
            } else if (MODE == MODE_UPD) {
                #pragma unroll
                for (int j = 0; j < 16; ++j)
                    vals[j] = (float)g.aggin[(size_t)trc * CDIM + kbase + j];
            } else { // MODE_ATTR, K = 128
                #pragma unroll
                for (int j = 0; j < 16; ++j)
                    vals[j] = g.emb[(size_t)rAtom * EMBD + kbase + j];
            }
            #pragma unroll
            for (int j = 0; j < 16; ++j)
                As[srow * 32 + seg + j] = (_Float16)vals[j];
        }
        __syncthreads();

        // A fragment per ISA 16-bit 16x32 layout
        const int arow = wave * 16 + (lane & 15);
        const int off0 = (lane >= 16) ? 8 : 0;
        union { v16h v; v8h h2[2]; } a;
        a.h2[0] = *(const v8h*)&As[arow * 32 + off0];
        a.h2[1] = *(const v8h*)&As[arow * 32 + off0 + 16];

        const _Float16* wpk = g.Wp + (size_t)ks * (16 * 32 * 16);
        if (ks + 1 < KPAD / 32)
            __builtin_prefetch(wpk + 16 * 32 * 16, 0, 1);

        #pragma unroll
        for (int n = 0; n < 16; ++n) {
            v16h b = *(const v16h*)(wpk + (size_t)((n * 32 + lane) * 16));
            acc[n] = __builtin_amdgcn_wmma_f32_16x16x32_f16(
                false, a.v, false, b, (short)0, acc[n], false, false);
        }
    }

    // epilogue: (row = base + j + 8*(lane>=16), col = n*16 + lane%16)
    const int rowb = r0 + wave * 16 + ((lane >> 4) << 3);
    const int ncol = lane & 15;
    #pragma unroll
    for (int n = 0; n < 16; ++n) {
        int col = n * 16 + ncol;
        float bb = g.bias[col];
        #pragma unroll
        for (int j = 0; j < 8; ++j) {
            int row = rowb + j;
            if (row >= g.nrows) continue;
            float v = acc[n][j] + bb;
            size_t o = (size_t)row * CDIM + col;
            if (MODE == MODE_NEO2) {
                g.h[o] = silu_f(v) * (float)g.grbf[o];
            } else if (MODE == MODE_UPD) {
                g.h[o] += silu_f(v);
            } else {
                g.outh[o] = (_Float16)silu_f(v);
            }
        }
    }
}

// ---------------- MSG GEMM fused with group-of-6 reduction ----------------
// 96 triplet rows/block (6 waves x 16-row tile) => exactly 16 agg rows (dst_e = t/6).
// Writes only agg[e0..e0+15] f16; the [T,256] message matrix never touches HBM.
__global__ __launch_bounds__(192) void msg_gemm_fused_kernel(GemmArgs g)
{
    __shared__ _Float16 As[96 * 32]  __attribute__((aligned(32)));   // 6 KB
    __shared__ _Float16 Ms[96 * 256] __attribute__((aligned(32)));   // 48 KB

    const int tid  = threadIdx.x;
    const int lane = tid & 31;
    const int wave = tid >> 5;           // 0..5
    const int r0   = blockIdx.x * 96;    // triplet row base (exact: T = 3750*96)

    v8f acc[16];
    {
        v8f z = {};
        #pragma unroll
        for (int n = 0; n < 16; ++n) acc[n] = z;
    }

    const int srow = tid >> 1;           // 0..95
    const int seg  = (tid & 1) << 4;
    const int tr   = r0 + srow;
    const int se   = g.src_e[tr];
    const int aj   = g.atom_j[tr];
    const float*    hrow   = g.h         + (size_t)se * CDIM;
    const _Float16* grow   = g.sbf_g     + (size_t)tr * CDIM;
    const _Float16* arow_p = g.attr_atom + (size_t)aj * CDIM;

    for (int ks = 0; ks < 8; ++ks) {
        __syncthreads();
        const int kbase = ks * 32 + seg;
        float vals[16];
        #pragma unroll
        for (int j = 0; j < 16; ++j) {
            int k = kbase + j;
            vals[j] = hrow[k] * (float)grow[k] + (float)arow_p[k];
        }
        #pragma unroll
        for (int j = 0; j < 16; ++j)
            As[srow * 32 + seg + j] = (_Float16)vals[j];
        __syncthreads();

        const int afrow = wave * 16 + (lane & 15);
        const int off0  = (lane >= 16) ? 8 : 0;
        union { v16h v; v8h h2[2]; } a;
        a.h2[0] = *(const v8h*)&As[afrow * 32 + off0];
        a.h2[1] = *(const v8h*)&As[afrow * 32 + off0 + 16];

        const _Float16* wpk = g.Wp + (size_t)ks * (16 * 32 * 16);
        if (ks + 1 < 8)
            __builtin_prefetch(wpk + 16 * 32 * 16, 0, 1);

        #pragma unroll
        for (int n = 0; n < 16; ++n) {
            v16h b = *(const v16h*)(wpk + (size_t)((n * 32 + lane) * 16));
            acc[n] = __builtin_amdgcn_wmma_f32_16x16x32_f16(
                false, a.v, false, b, (short)0, acc[n], false, false);
        }
    }

    // silu(m) -> LDS
    const int rowb = wave * 16 + ((lane >> 4) << 3);
    const int ncol = lane & 15;
    #pragma unroll
    for (int n = 0; n < 16; ++n) {
        int col = n * 16 + ncol;
        float bb = g.bias[col];
        #pragma unroll
        for (int j = 0; j < 8; ++j)
            Ms[(rowb + j) * 256 + col] = (_Float16)silu_f(acc[n][j] + bb);
    }
    __syncthreads();

    // reduce groups of 6 consecutive triplet rows -> agg edge rows
    const int e0 = blockIdx.x * 16;
    for (int idx = tid; idx < 16 * 256; idx += 192) {
        int ar = idx >> 8, c = idx & 255;
        float s = 0.0f;
        #pragma unroll
        for (int d = 0; d < DEG; ++d)
            s += (float)Ms[(ar * DEG + d) * 256 + c];
        g.outh[(size_t)(e0 + ar) * CDIM + c] = (_Float16)s;
    }
}

// ---------------- launcher ----------------
extern "C" void kernel_launch(void* const* d_in, const int* in_sizes, int n_in,
                              void* d_out, int out_size, void* d_ws, size_t ws_size,
                              hipStream_t stream)
{
    (void)in_sizes; (void)n_in; (void)out_size; (void)ws_size;

    const int*   x         = (const int*)  d_in[0];
    const float* pos       = (const float*)d_in[1];
    const int*   edge_idx  = (const int*)  d_in[2];
    const float* edge_attr = (const float*)d_in[3];
    const int*   neo_idx   = (const int*)  d_in[4];
    const int*   atom_i    = (const int*)  d_in[5];
    const int*   atom_j    = (const int*)  d_in[6];
    const int*   atom_k    = (const int*)  d_in[7];
    const float* emb_table = (const float*)d_in[8];
    const float* mat_W     = (const float*)d_in[9];
    const float* mat_b     = (const float*)d_in[10];
    const float* embt_W    = (const float*)d_in[11];
    const float* embt_b    = (const float*)d_in[12];
    const float* rbf_W     = (const float*)d_in[13];
    const float* rbf_b     = (const float*)d_in[14];
    const float* sbf_W     = (const float*)d_in[15];
    const float* sbf_b     = (const float*)d_in[16];
    const float* attr_W    = (const float*)d_in[17];
    const float* attr_b    = (const float*)d_in[18];
    const float* msg_W     = (const float*)d_in[19];
    const float* msg_b     = (const float*)d_in[20];
    const float* upd_W     = (const float*)d_in[21];
    const float* upd_b     = (const float*)d_in[22];
    const float* out_W     = (const float*)d_in[23];
    const float* out_b     = (const float*)d_in[24];
    float*       out       = (float*)d_out;

    // ---- carve workspace ----
    size_t off = 0;
    char* base = (char*)d_ws;
    auto carve = [&](size_t bytes) -> void* {
        void* p = base + off;
        off += (bytes + 255) & ~(size_t)255;
        return p;
    };
    float*     theta = (float*)    carve((size_t)NTRIP * 4);
    float*     env   = (float*)    carve((size_t)NEDGE * 4);
    float*     bvec  = (float*)    carve((size_t)NEDGE * 3 * 4);
    float*     nrbf  = (float*)    carve((size_t)NEDGE * NRBF * 4);
    _Float16*  grbf  = (_Float16*) carve((size_t)NEDGE * CDIM * 2);
    _Float16*  neo1  = (_Float16*) carve((size_t)NEDGE * CDIM * 2); // reused as agg
    float*     h     = (float*)    carve((size_t)NEDGE * CDIM * 4);
    _Float16*  attrA = (_Float16*) carve((size_t)NATOMS * CDIM * 2);
    _Float16*  sbfg  = (_Float16*) carve((size_t)NTRIP * CDIM * 2);
    float*     sbuf  = (float*)    carve((size_t)NEDGE * 4);
    _Float16*  matP  = (_Float16*) carve((size_t)352 * CDIM * 2);
    _Float16*  embtP = (_Float16*) carve((size_t)256 * CDIM * 2);
    _Float16*  sbfP  = (_Float16*) carve((size_t)128 * CDIM * 2);
    _Float16*  attrP = (_Float16*) carve((size_t)128 * CDIM * 2);
    _Float16*  msgP  = (_Float16*) carve((size_t)4 * 256 * CDIM * 2);
    _Float16*  updP  = (_Float16*) carve((size_t)4 * 256 * CDIM * 2);
    _Float16*  aggb  = neo1; // neo1 dead after NEO2

    const int* src_e = neo_idx;  // row 0 of neo_edge_index

    // ---- pack all GEMM weights into WMMA B-fragment layout (f16) ----
    pack_weights_kernel<<<352, 256, 0, stream>>>(mat_W,  matP,  338, 352);
    pack_weights_kernel<<<256, 256, 0, stream>>>(embt_W, embtP, 256, 256);
    pack_weights_kernel<<<128, 256, 0, stream>>>(sbf_W,  sbfP,  112, 128);
    pack_weights_kernel<<<128, 256, 0, stream>>>(attr_W, attrP, 128, 128);
    for (int l = 0; l < 4; ++l) {
        pack_weights_kernel<<<256, 256, 0, stream>>>(msg_W + (size_t)l * 256 * CDIM,
                                                     msgP + (size_t)l * 256 * CDIM, 256, 256);
        pack_weights_kernel<<<256, 256, 0, stream>>>(upd_W + (size_t)l * 256 * CDIM,
                                                     updP + (size_t)l * 256 * CDIM, 256, 256);
    }

    // ---- geometry / bases ----
    geom_kernel<<<(NEDGE + 255) / 256, 256, 0, stream>>>(pos, edge_idx, bvec, env, nrbf);
    grbf_kernel<<<NEDGE, 256, 0, stream>>>(nrbf, rbf_W, rbf_b, grbf);
    theta_kernel<<<(NTRIP + 255) / 256, 256, 0, stream>>>(pos, atom_i, atom_j, atom_k, theta);

    GemmArgs g = {};
    g.edge_attr = edge_attr; g.env = env; g.neo1 = neo1; g.grbf = grbf; g.h = h;
    g.src_e = src_e; g.atom_j = atom_j; g.sbf_g = sbfg; g.attr_atom = attrA;
    g.theta = theta; g.node_rbf = nrbf; g.aggin = aggb; g.emb = emb_table; g.xz = x;

    // attr_atom = silu(emb[x] @ attr_W + b)            [N,256]
    g.Wp = attrP; g.bias = attr_b; g.nrows = NATOMS; g.outh = attrA;
    gemm_wmma_kernel<MODE_ATTR, 128><<<(NATOMS + 127) / 128, 256, 0, stream>>>(g);

    // neo1 = silu((edge_attr*env) @ mat_W + b)         [E,256]
    g.Wp = matP; g.bias = mat_b; g.nrows = NEDGE; g.outh = neo1;
    gemm_wmma_kernel<MODE_NEO1, 352><<<(NEDGE + 127) / 128, 256, 0, stream>>>(g);

    // h = silu(neo1 @ embt_W + b) * grbf               [E,256] f32
    g.Wp = embtP; g.bias = embt_b; g.nrows = NEDGE;
    gemm_wmma_kernel<MODE_NEO2, 256><<<(NEDGE + 127) / 128, 256, 0, stream>>>(g);

    // sbf_g = silu(edge_sbf @ sbf_W + b)               [T,256]
    g.Wp = sbfP; g.bias = sbf_b; g.nrows = NTRIP; g.outh = sbfg;
    gemm_wmma_kernel<MODE_SBF, 128><<<(NTRIP + 127) / 128, 256, 0, stream>>>(g);

    // ---- 4 message-passing layers ----
    for (int l = 0; l < 4; ++l) {
        // agg = segsum6(silu((h[src]*sbf_g + attr_g) @ msg_W + b))   [E,256] fused
        g.Wp = msgP + (size_t)l * 256 * CDIM; g.bias = msg_b + l * CDIM;
        g.nrows = NTRIP; g.outh = aggb;
        msg_gemm_fused_kernel<<<NTRIP / 96, 192, 0, stream>>>(g);

        // h += silu(agg @ upd_W + b)                   [E,256]
        g.Wp = updP + (size_t)l * 256 * CDIM; g.bias = upd_b + l * CDIM;
        g.nrows = NEDGE;
        gemm_wmma_kernel<MODE_UPD, 256><<<(NEDGE + 127) / 128, 256, 0, stream>>>(g);
    }

    // ---- readout ----
    s_kernel<<<(NEDGE + 7) / 8, 256, 0, stream>>>(h, out_W, out_b, sbuf);
    out_kernel<<<(NATOMS + 255) / 256, 256, 0, stream>>>(pos, sbuf, bvec, out);
}
